// Encoder_28776280883967
// MI455X (gfx1250) — matrix operand order repaired and checked by reading, not compiled, not run on hardware
//
#include <hip/hip_runtime.h>
#include <hip/hip_bf16.h>

typedef __attribute__((ext_vector_type(16))) __bf16 v16bf;
typedef __attribute__((ext_vector_type(8)))  __bf16 v8bf;
typedef __attribute__((ext_vector_type(8)))  float  v8f;
typedef __attribute__((ext_vector_type(4)))  float  v4f;

#define LSEQ  256
#define BATCH 64
#define EMBD  512
#define HD    512

// ---------------------------------------------------------------------------
// Embedding gather + split into bf16 hi/lo (x = hi + lo, hi = bf16(x))
// ---------------------------------------------------------------------------
__global__ void embed_split_kernel(const int* __restrict__ tok,
                                   const float* __restrict__ emb,
                                   __bf16* __restrict__ hi,
                                   __bf16* __restrict__ lo) {
    int idx = blockIdx.x * 256 + threadIdx.x;      // over 16384*512
    int row = idx >> 9;                            // l*B + b
    int e   = idx & 511;
    int t   = tok[row];
    float v = emb[(long)t * EMBD + e];
    __bf16 h = (__bf16)v;
    __bf16 l = (__bf16)(v - (float)h);
    hi[idx] = h;
    lo[idx] = l;
}

// ---------------------------------------------------------------------------
// Generic f32 -> bf16 hi/lo split (for layer-1 activations)
// ---------------------------------------------------------------------------
__global__ void split_f32_kernel(const float* __restrict__ x,
                                 __bf16* __restrict__ hi,
                                 __bf16* __restrict__ lo, int n) {
    int idx = blockIdx.x * 256 + threadIdx.x;
    if (idx >= n) return;
    float v = x[idx];
    __bf16 h = (__bf16)v;
    hi[idx] = h;
    lo[idx] = (__bf16)(v - (float)h);
}

// ---------------------------------------------------------------------------
// Pack W (K,N) f32 row-major into W^T (N,K) bf16 hi/lo for WMMA B operand
// ---------------------------------------------------------------------------
__global__ void pack_wt_kernel(const float* __restrict__ W,
                               __bf16* __restrict__ hi,
                               __bf16* __restrict__ lo, int K, int N) {
    int idx = blockIdx.x * 256 + threadIdx.x;
    if (idx >= K * N) return;
    int n = idx / K;
    int k = idx - n * K;
    float v = W[(long)k * N + n];
    __bf16 h = (__bf16)v;
    hi[idx] = h;
    lo[idx] = (__bf16)(v - (float)h);
}

// ---------------------------------------------------------------------------
// GEMM: C(M,N) f32 = A(M,K) * B(K,N), A/B given as bf16 hi/lo pairs,
// B pre-packed as B^T (N,K). 3-pass split product for ~f32 accuracy.
// One wave -> 16x64 output tile. 256 threads = 8 waves per block.
// ---------------------------------------------------------------------------
__global__ __launch_bounds__(256) void gemm_bf16_split_kernel(
    const __bf16* __restrict__ Ahi, const __bf16* __restrict__ Alo,
    const __bf16* __restrict__ Bhi, const __bf16* __restrict__ Blo,
    float* __restrict__ C, int M, int N, int K) {
    const int lane  = threadIdx.x & 31;
    const int wave  = threadIdx.x >> 5;
    const int wg    = blockIdx.x * 8 + wave;
    const int tilesM = M >> 4;
    const int tm    = wg % tilesM;          // 16-row tile
    const int tn64  = wg / tilesM;          // 64-col tile
    const int ml    = lane & 15;            // m (A rows) or n (B cols) in tile
    const int hh    = lane >> 4;            // lane-half selector

    const __bf16* arow_hi = Ahi + (long)(tm * 16 + ml) * K;
    const __bf16* arow_lo = Alo + (long)(tm * 16 + ml) * K;

    v8f acc[4] = {};

    for (int kb = 0; kb < K; kb += 32) {
        // A tile (16x32): contiguous 16-elem run at K offset h*16
        v8bf a0h = *(const v8bf*)(arow_hi + kb + hh * 16);
        v8bf a1h = *(const v8bf*)(arow_hi + kb + hh * 16 + 8);
        v8bf a0l = *(const v8bf*)(arow_lo + kb + hh * 16);
        v8bf a1l = *(const v8bf*)(arow_lo + kb + hh * 16 + 8);
        v16bf a_hi = __builtin_shufflevector(a0h, a1h,
            0,1,2,3,4,5,6,7,8,9,10,11,12,13,14,15);
        v16bf a_lo = __builtin_shufflevector(a0l, a1l,
            0,1,2,3,4,5,6,7,8,9,10,11,12,13,14,15);

#pragma unroll
        for (int nt = 0; nt < 4; ++nt) {
            long boff = (long)(tn64 * 64 + nt * 16 + ml) * K + kb + hh * 16;
            v16bf b_hi = *(const v16bf*)(Bhi + boff);
            v16bf b_lo = *(const v16bf*)(Blo + boff);
            acc[nt] = __builtin_amdgcn_wmma_f32_16x16x32_bf16(
                false, a_hi, false, b_hi, (short)0, acc[nt], false, false);
            acc[nt] = __builtin_amdgcn_wmma_f32_16x16x32_bf16(
                false, a_hi, false, b_lo, (short)0, acc[nt], false, false);
            acc[nt] = __builtin_amdgcn_wmma_f32_16x16x32_bf16(
                false, a_lo, false, b_hi, (short)0, acc[nt], false, false);
        }
    }

    // D layout: VGPR r -> row r + 8*h; col = lane&15
#pragma unroll
    for (int nt = 0; nt < 4; ++nt) {
        int ncol = tn64 * 64 + nt * 16 + ml;
#pragma unroll
        for (int r = 0; r < 8; ++r) {
            C[(long)(tm * 16 + r + 8 * hh) * N + ncol] = acc[nt][r];
        }
    }
}

// ---------------------------------------------------------------------------
// Bidirectional SRU scan. One thread per (b, dir, n) recurrence chain.
// dir 0 walks t forward, dir 1 walks t backward (this realizes _flip_bwd /
// un-flip exactly). KD=4: residual from u[...,3]; KD=3: residual from x.
// ---------------------------------------------------------------------------
template <int KD>
__global__ void sru_scan_kernel(const float* __restrict__ u,
                                const float* __restrict__ bias,   // (2,2,H)
                                const float* __restrict__ xres,   // (L,B,2H) for KD==3
                                float* __restrict__ h,            // (L,B,2H)
                                float* __restrict__ c_out) {      // (B,2H)
    int idx = blockIdx.x * 256 + threadIdx.x;      // 65536 chains
    int n = idx & (HD - 1);
    int d = (idx >> 9) & 1;
    int b = idx >> 10;
    float bf = bias[0 * 2 * HD + d * HD + n];
    float br = bias[1 * 2 * HD + d * HD + n];
    const long colbase = (long)(d * HD + n) * KD;
    const long rw = (long)2 * HD * KD;             // u row width
    float c = 0.0f;
    int t  = (d == 0) ? 0 : (LSEQ - 1);
    int dt = (d == 0) ? 1 : -1;
    for (int l = 0; l < LSEQ; ++l, t += dt) {
        long row = (long)t * BATCH + b;
        const float* up = u + row * rw + colbase;
        float xt, fz, rz, xr;
        if (KD == 4) {
            v4f uv = *(const v4f*)up;
            xt = uv[0]; fz = uv[1]; rz = uv[2]; xr = uv[3];
        } else {
            xt = up[0]; fz = up[1]; rz = up[2];
            xr = xres[row * (2 * HD) + d * HD + n];
        }
        float f = 1.0f / (1.0f + __expf(-(fz + bf)));
        float r = 1.0f / (1.0f + __expf(-(rz + br)));
        c = f * c + (1.0f - f) * xt;
        h[row * (2 * HD) + d * HD + n] = r * tanhf(c) + (1.0f - r) * xr;
    }
    c_out[b * (2 * HD) + d * HD + n] = c;
}

// ---------------------------------------------------------------------------
// Wd (H, 2H) -> WdT (2H, H) for coalesced dense head
// ---------------------------------------------------------------------------
__global__ void wdt_kernel(const float* __restrict__ Wd, float* __restrict__ WdT) {
    int idx = blockIdx.x * 256 + threadIdx.x;      // 1024*512
    int k = idx >> 9;
    int m = idx & 511;
    WdT[idx] = Wd[(long)m * 1024 + k];
}

// ---------------------------------------------------------------------------
// hidden[i][b][m] = bd[m] + sum_k c_i[b][k] * Wd[m][k]   (tiny GEMM)
// ---------------------------------------------------------------------------
__global__ void dense_kernel(const float* __restrict__ c1,
                             const float* __restrict__ c2,
                             const float* __restrict__ WdT,
                             const float* __restrict__ bd,
                             float* __restrict__ out) {
    int idx = blockIdx.x * 256 + threadIdx.x;      // 2*64*512
    int m = idx & 511;
    int b = (idx >> 9) & 63;
    int i = idx >> 15;
    const float* cc = (i == 0) ? c1 : c2;
    float acc = bd[m];
    for (int k = 0; k < 2 * HD; ++k)
        acc += cc[b * (2 * HD) + k] * WdT[k * HD + m];
    out[idx] = acc;
}

// ---------------------------------------------------------------------------
extern "C" void kernel_launch(void* const* d_in, const int* in_sizes, int n_in,
                              void* d_out, int out_size, void* d_ws, size_t ws_size,
                              hipStream_t stream) {
    const int*   tok = (const int*)  d_in[0];
    const float* emb = (const float*)d_in[1];
    const float* W0  = (const float*)d_in[2];
    const float* b0  = (const float*)d_in[3];
    const float* W1  = (const float*)d_in[4];
    const float* b1  = (const float*)d_in[5];
    const float* Wd  = (const float*)d_in[6];
    const float* bd  = (const float*)d_in[7];

    float* out_h2     = (float*)d_out;                            // (L,B,2H)
    float* out_hidden = out_h2 + (size_t)LSEQ * BATCH * 2 * HD;   // (2,B,H)

    const int M = LSEQ * BATCH;          // 16384

    char* ws = (char*)d_ws;
    size_t off = 0;
    auto take = [&](size_t bytes) -> char* {
        char* p = ws + off;
        off += (bytes + 255) & ~(size_t)255;
        return p;
    };
    __bf16* aHi = (__bf16*)take((size_t)M * 1024 * 2);            // max K=1024
    __bf16* aLo = (__bf16*)take((size_t)M * 1024 * 2);
    __bf16* wHi = (__bf16*)take((size_t)3072 * 1024 * 2);         // max N*K
    __bf16* wLo = (__bf16*)take((size_t)3072 * 1024 * 2);
    float*  u   = (float*) take((size_t)M * 4096 * 4);            // max N=4096
    float*  h1  = (float*) take((size_t)M * 1024 * 4);
    float*  c1  = (float*) take((size_t)BATCH * 1024 * 4);
    float*  c2  = (float*) take((size_t)BATCH * 1024 * 4);
    float*  WdT = (float*) take((size_t)1024 * 512 * 4);

    // ---- layer 0: x = emb[tok]; u0 = x @ W0 ; scan (k=4) ----
    embed_split_kernel<<<(M * EMBD) / 256, 256, 0, stream>>>(tok, emb, aHi, aLo);
    pack_wt_kernel<<<(EMBD * 4096 + 255) / 256, 256, 0, stream>>>(W0, wHi, wLo, EMBD, 4096);
    {
        int waves = (M / 16) * (4096 / 64);
        gemm_bf16_split_kernel<<<waves / 8, 256, 0, stream>>>(
            aHi, aLo, wHi, wLo, u, M, 4096, EMBD);
    }
    sru_scan_kernel<4><<<(BATCH * 2 * HD) / 256, 256, 0, stream>>>(u, b0, nullptr, h1, c1);

    // ---- layer 1: u1 = h1 @ W1 ; scan (k=3) ----
    split_f32_kernel<<<(M * 1024) / 256, 256, 0, stream>>>(h1, aHi, aLo, M * 1024);
    pack_wt_kernel<<<(1024 * 3072 + 255) / 256, 256, 0, stream>>>(W1, wHi, wLo, 1024, 3072);
    {
        int waves = (M / 16) * (3072 / 64);
        gemm_bf16_split_kernel<<<waves / 8, 256, 0, stream>>>(
            aHi, aLo, wHi, wLo, u, M, 3072, 1024);
    }
    sru_scan_kernel<3><<<(BATCH * 2 * HD) / 256, 256, 0, stream>>>(u, b1, h1, out_h2, c2);

    // ---- dense head: hidden = stack(c1,c2) @ Wd^T + bd ----
    wdt_kernel<<<(1024 * 512) / 256, 256, 0, stream>>>(Wd, WdT);
    dense_kernel<<<(2 * BATCH * HD) / 256, 256, 0, stream>>>(c1, c2, WdT, bd, out_hidden);
}
